// SuperpositionRouter_56642028699743
// MI455X (gfx1250) — compile-verified, matrix operand order checked by reading, not run on hardware
//
#include <hip/hip_runtime.h>

typedef __attribute__((ext_vector_type(16))) __bf16        v16bf;
typedef __attribute__((ext_vector_type(8)))  float         v8f;
typedef __attribute__((ext_vector_type(4)))  float         v4f;
typedef __attribute__((ext_vector_type(4)))  unsigned int  v4u;
typedef __attribute__((ext_vector_type(4)))  int           v4i;
typedef __attribute__((ext_vector_type(2)))  unsigned int  v2u;

#define MTOT 16384   // 4 * 4096 rows of x
#define NTOT 512     // tiles
#define KTOT 2048    // feature dim
#define BM   128
#define BN   64
#define BK   64
#define LDA  72      // padded LDS row stride (bf16 elems): 144B, 16B-aligned

// gfx1250 async global->LDS copy (ASYNCcnt-tracked), if the toolchain exposes it
#if defined(__has_builtin)
#  if __has_builtin(__builtin_amdgcn_global_load_async_to_lds_b128) && \
      __has_builtin(__builtin_amdgcn_s_wait_asynccnt)
#    define USE_ASYNC_LDS 1
#  endif
#endif

__device__ __forceinline__ unsigned short f32_to_bf16(float f) {
  unsigned int u = __float_as_uint(f);
  u += 0x7fffu + ((u >> 16) & 1u);          // round-to-nearest-even
  return (unsigned short)(u >> 16);
}
__device__ __forceinline__ float bf16_to_f32(unsigned short h) {
  return __uint_as_float(((unsigned int)h) << 16);
}

// ---------------------------------------------------------------------------
// Kernel 1: ternarize signatures -> bf16 (+/-1, 0 are exact), inv-norm = 1/sqrt(nnz)
// ---------------------------------------------------------------------------
__global__ __launch_bounds__(256) void prep_sigs(const float* __restrict__ sig,
                                                 unsigned short* __restrict__ bs,
                                                 float* __restrict__ invs) {
  const int row = blockIdx.x;
  const int t   = threadIdx.x;
  const float* src = sig + (size_t)row * KTOT;
  unsigned short* dst = bs + (size_t)row * KTOT;
  int nnz = 0;
  for (int k = t; k < KTOT; k += 256) {
    float v = src[k];
    unsigned short s = (v > 0.f) ? 0x3F80u : ((v < 0.f) ? 0xBF80u : 0u);
    nnz += (v != 0.f);
    dst[k] = s;
  }
  __shared__ int red[256];
  red[t] = nnz; __syncthreads();
  for (int off = 128; off > 0; off >>= 1) {
    if (t < off) red[t] += red[t + off];
    __syncthreads();
  }
  if (t == 0) invs[row] = 1.0f / fmaxf(sqrtf((float)red[0]), 1e-12f);
}

// ---------------------------------------------------------------------------
// Kernel 2: per-row inverse L2 norm of x (f32, b128 reads)
// ---------------------------------------------------------------------------
__global__ __launch_bounds__(256) void xnorms(const float* __restrict__ x,
                                              float* __restrict__ invx) {
  const int row = blockIdx.x;
  const int t   = threadIdx.x;
  const v4f* src = (const v4f*)(x + (size_t)row * KTOT);
  float ss = 0.f;
  for (int k = t; k < KTOT / 4; k += 256) {
    v4f v = src[k];
    ss += v.x * v.x + v.y * v.y + v.z * v.z + v.w * v.w;
  }
  __shared__ float red[256];
  red[t] = ss; __syncthreads();
  for (int off = 128; off > 0; off >>= 1) {
    if (t < off) red[t] += red[t + off];
    __syncthreads();
  }
  if (t == 0) invx[row] = 1.0f / fmaxf(sqrtf(red[0]), 1e-12f);
}

// ---------------------------------------------------------------------------
// Kernel 3: WMMA bf16 split-precision GEMM with fused normalization epilogue
//   scores[m, n] = (x[m,:] . sign(sig[n,:])) * invx[m] * invs[n]
// Block = 256 threads = 8 waves; tile BM x BN, K-step BK.
// Wave w: rows [w*16, w*16+16) of the block tile, all 4 N-subtiles.
// ---------------------------------------------------------------------------
__global__ __launch_bounds__(256)
void gemm_scores(const float* __restrict__ X,
                 const unsigned short* __restrict__ Bmat,
                 const float* __restrict__ invx,
                 const float* __restrict__ invs,
                 float* __restrict__ scores) {
  __shared__ unsigned short Ahi[BM * LDA];   // 18 KB
  __shared__ unsigned short Alo[BM * LDA];   // 18 KB
  __shared__ unsigned short Bs [BN * LDA];   // 9 KB

  const int bx   = blockIdx.x;          // N tile: 0..7
  const int by   = blockIdx.y;          // M tile: 0..127
  const int t    = threadIdx.x;
  const int wave = t >> 5;
  const int lane = t & 31;
  const int hlf  = lane >> 4;           // 0: lanes 0-15, 1: lanes 16-31
  const int mn   = lane & 15;           // row (A) / col (B) within 16x16 tile

  const int row0 = by * BM;
  const int col0 = bx * BN;

  v8f acc[4];
#pragma unroll
  for (int i = 0; i < 4; ++i) acc[i] = (v8f){0.f,0.f,0.f,0.f,0.f,0.f,0.f,0.f};

  for (int k0 = 0; k0 < KTOT; k0 += BK) {
    __syncthreads();   // protect LDS from previous iteration's consumers

    // ---- B tile first (async copy if available): 64x64 bf16, 2 b128-chunks/thread
#pragma unroll
    for (int i = 0; i < 2; ++i) {
      const int c  = t + i * 256;       // 0..511
      const int br = c >> 3;            // 0..63
      const int bc = (c & 7) << 3;      // 0..56
      const unsigned short* gp = Bmat + (size_t)(col0 + br) * KTOT + k0 + bc;
#if defined(USE_ASYNC_LDS)
      __builtin_amdgcn_global_load_async_to_lds_b128(
          (__attribute__((address_space(1))) v4i*)(void*)gp,
          (__attribute__((address_space(3))) v4i*)(void*)&Bs[br * LDA + bc],
          0, 0);
#else
      const v4u v = *(const v4u*)gp;
      *(v4u*)&Bs[br * LDA + bc] = v;
#endif
    }

    // ---- load A tile (f32 -> split hi/lo bf16): 128x64, 8 float4-chunks/thread
#pragma unroll
    for (int i = 0; i < 8; ++i) {
      const int c  = t + i * 256;       // 0..2047
      const int ar = c >> 4;            // 0..127
      const int ac = (c & 15) << 2;     // 0..60
      const float* gp = X + (size_t)(row0 + ar) * KTOT + k0 + ac;
      if (k0 + BK < KTOT) __builtin_prefetch((const void*)(gp + BK), 0, 1);
      const v4f v = *(const v4f*)gp;
      unsigned short h0 = f32_to_bf16(v.x), h1 = f32_to_bf16(v.y);
      unsigned short h2 = f32_to_bf16(v.z), h3 = f32_to_bf16(v.w);
      unsigned short l0 = f32_to_bf16(v.x - bf16_to_f32(h0));
      unsigned short l1 = f32_to_bf16(v.y - bf16_to_f32(h1));
      unsigned short l2 = f32_to_bf16(v.z - bf16_to_f32(h2));
      unsigned short l3 = f32_to_bf16(v.w - bf16_to_f32(h3));
      v2u ph, pl;
      ph.x = (unsigned)h0 | ((unsigned)h1 << 16);
      ph.y = (unsigned)h2 | ((unsigned)h3 << 16);
      pl.x = (unsigned)l0 | ((unsigned)l1 << 16);
      pl.y = (unsigned)l2 | ((unsigned)l3 << 16);
      *(v2u*)&Ahi[ar * LDA + ac] = ph;
      *(v2u*)&Alo[ar * LDA + ac] = pl;
    }

#if defined(USE_ASYNC_LDS)
    __builtin_amdgcn_s_wait_asynccnt(0);  // drain this wave's async LDS writes
#endif
    __syncthreads();

    // ---- compute: two 32-wide K steps per stage
#pragma unroll
    for (int kk = 0; kk < BK; kk += 32) {
      // A fragment layout (16-bit, 16x32): VGPR0-3 = k in [kk+8*hlf, +8),
      // VGPR4-7 = k in [kk+16+8*hlf, +8)  -> two 16B LDS loads each.
      union Frag { v16bf v; v4u q[2]; };
      Frag ah, al;
      const int abase = (wave * 16 + mn) * LDA + kk + 8 * hlf;
      ah.q[0] = *(const v4u*)&Ahi[abase];
      ah.q[1] = *(const v4u*)&Ahi[abase + 16];
      al.q[0] = *(const v4u*)&Alo[abase];
      al.q[1] = *(const v4u*)&Alo[abase + 16];
#pragma unroll
      for (int nt = 0; nt < 4; ++nt) {
        Frag bf;
        const int bbase = (nt * 16 + mn) * LDA + kk + 8 * hlf;
        bf.q[0] = *(const v4u*)&Bs[bbase];
        bf.q[1] = *(const v4u*)&Bs[bbase + 16];
        acc[nt] = __builtin_amdgcn_wmma_f32_16x16x32_bf16(
            false, ah.v, false, bf.v, (short)0, acc[nt], false, false);
        acc[nt] = __builtin_amdgcn_wmma_f32_16x16x32_bf16(
            false, al.v, false, bf.v, (short)0, acc[nt], false, false);
      }
    }
  }

  // ---- epilogue: scale by 1/||x|| and 1/sqrt(nnz), store f32 scores
  const float* ivx = invx + row0 + wave * 16;
  float ivn[4];
#pragma unroll
  for (int nt = 0; nt < 4; ++nt) ivn[nt] = invs[col0 + nt * 16 + mn];
#pragma unroll
  for (int nt = 0; nt < 4; ++nt) {
#pragma unroll
    for (int r = 0; r < 8; ++r) {
      const int m = r + 8 * hlf;                 // C/D layout: VGPR r -> M = r + 8*half
      const float sc = acc[nt][r] * ivx[m] * ivn[nt];
      scores[(size_t)(row0 + wave * 16 + m) * NTOT + col0 + nt * 16 + mn] = sc;
    }
  }
}

// ---------------------------------------------------------------------------
// Kernel 4: row-wise argmax over 512 scores, first-index tie-break
// ---------------------------------------------------------------------------
__global__ __launch_bounds__(256) void argmax_rows(const float* __restrict__ scores,
                                                   float* __restrict__ idx_out) {
  const int row = blockIdx.x;
  const int t   = threadIdx.x;
  const float* s = scores + (size_t)row * NTOT;
  float v0 = s[t], v1 = s[t + 256];
  float bv; int bi;
  if (v1 > v0) { bv = v1; bi = t + 256; } else { bv = v0; bi = t; }
  __shared__ float sv[256];
  __shared__ int   si[256];
  sv[t] = bv; si[t] = bi; __syncthreads();
  for (int off = 128; off > 0; off >>= 1) {
    if (t < off) {
      float ov = sv[t + off]; int oi = si[t + off];
      if (ov > sv[t] || (ov == sv[t] && oi < si[t])) { sv[t] = ov; si[t] = oi; }
    }
    __syncthreads();
  }
  if (t == 0) idx_out[row] = (float)si[0];
}

// ---------------------------------------------------------------------------
extern "C" void kernel_launch(void* const* d_in, const int* in_sizes, int n_in,
                              void* d_out, int out_size, void* d_ws, size_t ws_size,
                              hipStream_t stream) {
  const float* x   = (const float*)d_in[0];   // [4, 4096, 2048]
  const float* sig = (const float*)d_in[1];   // [512, 2048]

  float* out      = (float*)d_out;
  float* tile_idx = out;                      // 16384 floats (argmax indices)
  float* scores   = out + MTOT;               // 16384*512 floats

  char* ws = (char*)d_ws;
  unsigned short* Bbf = (unsigned short*)ws;                        // 2 MB
  float* invs = (float*)(ws + (size_t)NTOT * KTOT * sizeof(unsigned short));
  float* invx = invs + NTOT;                                       // +64 KB

  prep_sigs<<<NTOT, 256, 0, stream>>>(sig, Bbf, invs);
  xnorms  <<<MTOT, 256, 0, stream>>>(x, invx);

  dim3 grid(NTOT / BN, MTOT / BM, 1);
  gemm_scores<<<grid, 256, 0, stream>>>(x, Bbf, invx, invs, scores);

  argmax_rows<<<MTOT, 256, 0, stream>>>(scores, tile_idx);
}
